// GATLayer_67577015435505
// MI455X (gfx1250) — matrix operand order repaired and checked by reading, not compiled, not run on hardware
//
#include <hip/hip_runtime.h>
#include <hip/hip_bf16.h>
#include <stdint.h>

typedef __attribute__((ext_vector_type(16))) _Float16 v16h;
typedef __attribute__((ext_vector_type(8)))  _Float16 v8h;
typedef __attribute__((ext_vector_type(8)))  float    v8f;

#define NNODES  4096
#define NEDGES  131072
#define IN_DIMC 1024
#define OUT_DIMC 512
#define NHEAD   4

// Low 32 bits of a generic LDS pointer are the wave-relative LDS byte offset
// (ISA 10.2, LDS aperture).
static __device__ __forceinline__ unsigned lds_off(const void* p) {
  return (unsigned)(uintptr_t)p;
}

// Async DMA: LDS[vdst] = MEM[vaddr], 16B per lane, tracked by ASYNCcnt.
static __device__ __forceinline__ void async_copy_b128(unsigned lds_byte_off,
                                                       const void* gaddr) {
  asm volatile("global_load_async_to_lds_b128 %0, %1, off"
               :: "v"(lds_byte_off), "v"(gaddr)
               : "memory");
}

// ---------------------------------------------------------------------------
// Utility kernels
// ---------------------------------------------------------------------------
__global__ void zero_f32(float* __restrict__ p, int n) {
  int i = blockIdx.x * blockDim.x + threadIdx.x;
  int stride = gridDim.x * blockDim.x;
  for (; i < n; i += stride) p[i] = 0.0f;
}

__global__ void cvt_f32_f16(const float* __restrict__ src, _Float16* __restrict__ dst, int n) {
  int i = blockIdx.x * blockDim.x + threadIdx.x;
  if (i < n) dst[i] = (_Float16)src[i];
}

// W: [H][IN][OUT] f32  ->  Wt: [H][OUT][IN] f16  (GEMM B-tiles contiguous in K)
__global__ void cvtT_W(const float* __restrict__ W, _Float16* __restrict__ Wt) {
  int gid = blockIdx.x * blockDim.x + threadIdx.x;       // h*IN*OUT + k*OUT + o
  int o = gid % OUT_DIMC;
  int k = (gid / OUT_DIMC) % IN_DIMC;
  int h = gid / (OUT_DIMC * IN_DIMC);
  Wt[((size_t)h * OUT_DIMC + o) * IN_DIMC + k] = (_Float16)W[gid];
}

// ---------------------------------------------------------------------------
// GEMM: h[hd] = x(4096x1024) @ W[hd](1024x512), f16 WMMA with f32 accum.
// Block = 256 threads (8 waves); block tile 128(M) x 128(N); wave tile 32x64.
// Triple-buffered LDS filled by async global->LDS DMA (ASYNCcnt), one
// workgroup barrier per k-step, 8 WMMAs per wave per k-step.
// ---------------------------------------------------------------------------
__global__ __launch_bounds__(256) void gat_gemm(
    const _Float16* __restrict__ A,   // [NNODES][IN] row-major
    const _Float16* __restrict__ Bt,  // [H][OUT][IN] row-major (pre-transposed)
    float* __restrict__ C)            // [H][NNODES][OUT]
{
  __shared__ _Float16 sA[3][128][40];   // 32 K-halves + 8 pad, 3 pipeline stages
  __shared__ _Float16 sB[3][128][40];

  const int h    = blockIdx.z;
  const int bm   = blockIdx.y * 128;
  const int bn   = blockIdx.x * 128;
  const int tid  = threadIdx.x;
  const int lane = tid & 31;
  const int wave = tid >> 5;
  const int wm   = wave & 3;            // row sub-block 0..3 (x32)
  const int wn   = wave >> 2;           // col sub-block 0..1 (x64)
  const int g    = lane >> 4;           // half-wave index
  const int l16  = lane & 15;

  const _Float16* Bh = Bt + (size_t)h * OUT_DIMC * IN_DIMC;

  // Loader mapping: 256 threads x 16B x 2 instrs = 8KB tile (128 rows x 32 halves)
  const int lrow = tid >> 2;            // 0..63
  const int loff = (tid & 3) * 8;       // 0 / 8 / 16 / 24 halves

  v8f acc[2][4] = {};

  const int KSTEPS = IN_DIMC / 32;      // 32

  // Stage issue: 4 async b128 copies per thread (A rows 0-63 / 64-127, B same)
  auto issue_stage = [&](int buf, int kk) {
    const _Float16* ga0 = &A [(size_t)(bm + lrow)      * IN_DIMC + kk + loff];
    const _Float16* ga1 = &A [(size_t)(bm + lrow + 64) * IN_DIMC + kk + loff];
    const _Float16* gb0 = &Bh[(size_t)(bn + lrow)      * IN_DIMC + kk + loff];
    const _Float16* gb1 = &Bh[(size_t)(bn + lrow + 64) * IN_DIMC + kk + loff];
    async_copy_b128(lds_off(&sA[buf][lrow]     [loff]), ga0);
    async_copy_b128(lds_off(&sA[buf][lrow + 64][loff]), ga1);
    async_copy_b128(lds_off(&sB[buf][lrow]     [loff]), gb0);
    async_copy_b128(lds_off(&sB[buf][lrow + 64][loff]), gb1);
  };

  issue_stage(0, 0);

  for (int ks = 0; ks < KSTEPS; ++ks) {
    const int cur = ks % 3;
    if (ks + 1 < KSTEPS) {
      issue_stage((ks + 1) % 3, (ks + 1) * 32);
      // own outstanding: stage ks (4) + stage ks+1 (4) -> retire stage ks
      asm volatile("s_wait_asynccnt %0" :: "n"(4));
    } else {
      asm volatile("s_wait_asynccnt %0" :: "n"(0));
    }
    __syncthreads();   // all waves' stage-ks DMA landed in LDS

    // A fragment: lane l16 = row m; halves K[g*8..g*8+7], K[16+g*8..16+g*8+7]
    v16h afrag[2];
#pragma unroll
    for (int mi = 0; mi < 2; ++mi) {
      int r = wm * 32 + mi * 16 + l16;
      v8h lo = *reinterpret_cast<const v8h*>(&sA[cur][r][g * 8]);
      v8h hi = *reinterpret_cast<const v8h*>(&sA[cur][r][16 + g * 8]);
      afrag[mi] = __builtin_shufflevector(lo, hi, 0,1,2,3,4,5,6,7,8,9,10,11,12,13,14,15);
    }
    // B fragment: lane l16 = col n; halves K[g*8..g*8+7], K[16+g*8..16+g*8+7]
    v16h bfrag[4];
#pragma unroll
    for (int ni = 0; ni < 4; ++ni) {
      int c = wn * 64 + ni * 16 + l16;
      v8h lo = *reinterpret_cast<const v8h*>(&sB[cur][c][g * 8]);
      v8h hi = *reinterpret_cast<const v8h*>(&sB[cur][c][16 + g * 8]);
      bfrag[ni] = __builtin_shufflevector(lo, hi, 0,1,2,3,4,5,6,7,8,9,10,11,12,13,14,15);
    }
#pragma unroll
    for (int mi = 0; mi < 2; ++mi)
#pragma unroll
      for (int ni = 0; ni < 4; ++ni)
        acc[mi][ni] = __builtin_amdgcn_wmma_f32_16x16x32_f16(
            false, afrag[mi], false, bfrag[ni], (short)0, acc[mi][ni], false, false);
    // No trailing barrier needed: the barrier at step ks+1 separates any
    // wave's DMA issue into buffer (ks+3)%3 from this step's reads (3 bufs).
  }

  // Store: C/D layout -> element v of v8f is (M = v + 8*g, N = l16)
  float* Ch = C + (size_t)h * NNODES * OUT_DIMC;
#pragma unroll
  for (int mi = 0; mi < 2; ++mi)
#pragma unroll
    for (int ni = 0; ni < 4; ++ni) {
      int col = bn + wn * 64 + ni * 16 + l16;
#pragma unroll
      for (int v = 0; v < 8; ++v) {
        int row = bm + wm * 32 + mi * 16 + v + 8 * g;
        Ch[(size_t)row * OUT_DIMC + col] = acc[mi][ni][v];
      }
    }
}

// ---------------------------------------------------------------------------
// Per-node attention logits: alpha_src/dst[h][n] = h[h,n,:] . a_{src,dst}[h]
// One wave per (h,n) row.
// ---------------------------------------------------------------------------
__global__ void gat_alpha(const float* __restrict__ hbuf, const float* __restrict__ a,
                          float* __restrict__ asrc, float* __restrict__ adst)
{
  int wave = threadIdx.x >> 5;
  int lane = threadIdx.x & 31;
  int rid  = blockIdx.x * 8 + wave;      // 0 .. H*N-1
  int h = rid / NNODES;
  int n = rid % NNODES;
  const float* hv = hbuf + ((size_t)h * NNODES + n) * OUT_DIMC;
  const float* ah = a + h * 2 * OUT_DIMC;
  float s = 0.0f, d = 0.0f;
  for (int o = lane; o < OUT_DIMC; o += 32) {
    float v = hv[o];
    s += v * ah[o];
    d += v * ah[OUT_DIMC + o];
  }
  for (int off = 16; off > 0; off >>= 1) {
    s += __shfl_xor(s, off, 32);
    d += __shfl_xor(d, off, 32);
  }
  if (lane == 0) { asrc[rid] = s; adst[rid] = d; }
}

// Column sums h_sum[h][o] = sum_n h[h][n][o]
__global__ void gat_hsum(const float* __restrict__ hbuf, float* __restrict__ hsum) {
  int h = blockIdx.x;
  int chunk = blockIdx.y;                // 16 chunks of 256 rows
  int o = threadIdx.x;                   // 512
  const float* base = hbuf + ((size_t)h * NNODES + chunk * 256) * OUT_DIMC + o;
  float s = 0.0f;
  for (int n = 0; n < 256; ++n) s += base[(size_t)n * OUT_DIMC];
  atomicAdd(&hsum[h * OUT_DIMC + o], s);
}

// ---------------------------------------------------------------------------
// Edge scatter: w = exp(leakyrelu(as[row]+ad[col])) - 1
//   denom[h][row] += w;  num[h][row][:] += w * h[h][col][:]
// One wave per (edge, head). num/denom are L2-resident (32MB << 192MB L2).
// ---------------------------------------------------------------------------
__global__ void gat_edge(const int* __restrict__ eidx, const float* __restrict__ hbuf,
                         const float* __restrict__ asrc, const float* __restrict__ adst,
                         float* __restrict__ num, float* __restrict__ denom)
{
  int wave = threadIdx.x >> 5;
  int lane = threadIdx.x & 31;
  unsigned wid = blockIdx.x * 8u + wave;   // 0 .. E*H-1
  int e = wid >> 2;                        // / NHEAD
  int h = wid & 3;
  int row = eidx[e];
  int col = eidx[NEDGES + e];
  float ev = asrc[h * NNODES + row] + adst[h * NNODES + col];
  ev = ev > 0.0f ? ev : 0.2f * ev;
  float w = __expf(ev) - 1.0f;
  if (lane == 0) atomicAdd(&denom[h * NNODES + row], w);
  const float* hc = hbuf + ((size_t)h * NNODES + col) * OUT_DIMC;
  float* nr = num + ((size_t)h * NNODES + row) * OUT_DIMC;
#pragma unroll
  for (int i = 0; i < OUT_DIMC / 32; ++i) {
    int o = lane + 32 * i;
    atomicAdd(&nr[o], w * hc[o]);
  }
}

// out[n][h*OUT+o] = (h_sum[h][o] + num[h][n][o]) / (N + denom[h][n])
__global__ void gat_final(const float* __restrict__ num, const float* __restrict__ denom,
                          const float* __restrict__ hsum, float* __restrict__ out)
{
  int gid = blockIdx.x * 256 + threadIdx.x;    // n*H*OUT + h*OUT + o
  int o = gid % OUT_DIMC;
  int h = (gid / OUT_DIMC) % NHEAD;
  int n = gid / (OUT_DIMC * NHEAD);
  float nv = hsum[h * OUT_DIMC + o] + num[((size_t)h * NNODES + n) * OUT_DIMC + o];
  float dv = (float)NNODES + denom[h * NNODES + n];
  out[gid] = nv / dv;
}

// ---------------------------------------------------------------------------
extern "C" void kernel_launch(void* const* d_in, const int* in_sizes, int n_in,
                              void* d_out, int out_size, void* d_ws, size_t ws_size,
                              hipStream_t stream) {
  const float* x    = (const float*)d_in[0];
  const int*   eidx = (const int*)d_in[1];     // (2, E)
  const float* W    = (const float*)d_in[2];   // (H, IN, OUT)
  const float* a    = (const float*)d_in[3];   // (H, 2*OUT)
  float* out = (float*)d_out;

  // Workspace layout
  const size_t HSZ = (size_t)NHEAD * NNODES * OUT_DIMC;   // 8,388,608
  float*    hbuf  = (float*)d_ws;
  float*    num   = hbuf + HSZ;
  float*    denom = num + HSZ;                            // 16384
  float*    hsum  = denom + (size_t)NHEAD * NNODES;       // 2048
  float*    asrc  = hsum + (size_t)NHEAD * OUT_DIMC;      // 16384
  float*    adst  = asrc + (size_t)NHEAD * NNODES;        // 16384
  _Float16* xh    = (_Float16*)(adst + (size_t)NHEAD * NNODES);
  _Float16* wt    = xh + (size_t)NNODES * IN_DIMC;        // 4,194,304 halves

  // Zero num + denom + hsum (contiguous)
  int nz = (int)(HSZ + NHEAD * NNODES + NHEAD * OUT_DIMC);
  zero_f32<<<4096, 256, 0, stream>>>(num, nz);

  // f32 -> f16 conversions (B pre-transposed to [h][o][k])
  cvt_f32_f16<<<(NNODES * IN_DIMC) / 256, 256, 0, stream>>>(x, xh, NNODES * IN_DIMC);
  cvtT_W<<<(NHEAD * IN_DIMC * OUT_DIMC) / 256, 256, 0, stream>>>(W, wt);

  // WMMA GEMM with async global->LDS pipeline
  dim3 ggrid(OUT_DIMC / 128, NNODES / 128, NHEAD);
  gat_gemm<<<ggrid, 256, 0, stream>>>(xh, wt, hbuf);

  // Attention pieces
  gat_alpha<<<(NHEAD * NNODES) / 8, 256, 0, stream>>>(hbuf, a, asrc, adst);
  gat_hsum<<<dim3(NHEAD, 16), 512, 0, stream>>>(hbuf, hsum);
  gat_edge<<<(NEDGES * NHEAD) / 8, 256, 0, stream>>>(eidx, hbuf, asrc, adst, num, denom);
  gat_final<<<(NNODES * NHEAD * OUT_DIMC) / 256, 256, 0, stream>>>(num, denom, hsum, out);
}
